// Controller_6828998001491
// MI455X (gfx1250) — matrix operand order repaired and checked by reading, not compiled, not run on hardware
//
#include <hip/hip_runtime.h>

// ---------------- problem constants ----------------
#define HSZ     2048          // LSTM hidden size
#define G4      8192          // 4*HSZ gate rows
#define KTOT    4096          // concat(x,h) length
#define KSPLIT  4             // K-split for GEMV parallelism
#define KC      (KTOT/KSPLIT) // 1024 K per wave
#define NC      12            // num choices
#define NLAYERS 32

typedef __attribute__((ext_vector_type(16))) __bf16 v16bf;
typedef __attribute__((ext_vector_type(8)))  __bf16 v8bf;
typedef __attribute__((ext_vector_type(8)))  float  v8f;

__device__ __forceinline__ unsigned short f2bf(float f) {
  unsigned u = __float_as_uint(f);
  u += 0x7FFFu + ((u >> 16) & 1u);   // round-to-nearest-even to bf16
  return (unsigned short)(u >> 16);
}

__device__ __forceinline__ float sigmoidf(float x) {
  return 1.0f / (1.0f + __expf(-x));
}

// ------------- one-time fp32 -> bf16 weight conversion (also warms L2) -------------
__global__ __launch_bounds__(256) void cvt_weights(const float* __restrict__ wih,
                                                   const float* __restrict__ whh,
                                                   unsigned short* __restrict__ wihB,
                                                   unsigned short* __restrict__ whhB) {
  unsigned i = blockIdx.x * 256u + threadIdx.x;   // grid sized exactly to 8192*2048
  wihB[i] = f2bf(wih[i]);
  whhB[i] = f2bf(whh[i]);
}

// ------------- per-call state init -------------
__global__ __launch_bounds__(256) void init_state(const float* __restrict__ prev_c,
                                                  const float* __restrict__ prev_h,
                                                  const float* __restrict__ encoder,
                                                  float* __restrict__ cstate,
                                                  unsigned short* __restrict__ xh,
                                                  float* __restrict__ out) {
  int t = threadIdx.x;
  for (int q = 0; q < HSZ / 256; ++q) {
    int j = t + 256 * q;
    cstate[j]    = prev_c[j];
    xh[j]        = f2bf(encoder[j]);   // x0 = encoder row 0
    xh[HSZ + j]  = f2bf(prev_h[j]);    // h0 = 0
  }
  if (t == 0) { out[NLAYERS] = 0.0f; out[NLAYERS + 1] = 0.0f; }
}

// ------------- WMMA GEMV: partial[kc][r] = sum_{k in chunk kc} Wcat[r,k] * xh[k] -------------
// A operand = xh chunk broadcast into matrix row 0 (lanes 0 and 16 per CDNA5 A layout).
// B operand = W^T tile: lane n holds W row (r0+n), 16 contiguous K values (32B load).
// D row 0 -> acc[0] on lanes 0..15 = 16 gate-row partials.
__global__ __launch_bounds__(256) void gemv_wmma(const unsigned short* __restrict__ wihB,
                                                 const unsigned short* __restrict__ whhB,
                                                 const unsigned short* __restrict__ xh,
                                                 float* __restrict__ partial) {
  const int wid      = (blockIdx.x * 256 + threadIdx.x) >> 5;  // 0..2047
  const int lane     = threadIdx.x & 31;
  const int row_tile = wid >> 2;            // 0..511
  const int kchunk   = wid & (KSPLIT - 1);  // 0..3
  const int r0       = row_tile << 4;
  const int kbase    = kchunk * KC;

  const unsigned short* W = (kbase < HSZ) ? wihB : whhB;
  const int kmod   = kbase & (HSZ - 1);     // K offset inside selected matrix
  const int colSel = lane & 15;             // which W row this lane's B column holds
  const int khalf  = (lane >> 4) << 4;      // B: lanes<16 -> K 0..15, lanes>=16 -> K 16..31
  const int aoff   = (lane >> 4) << 4;      // A: lanes<16 -> K 0..15; lanes>=16 -> K 16..31

  const __bf16* Wrow = (const __bf16*)W + (unsigned)(r0 + colSel) * HSZ + kmod + khalf;
  const __bf16* xv   = (const __bf16*)xh + kbase;

  v8f acc = {};
  #pragma unroll 4
  for (int kk = 0; kk < KC; kk += 32) {
    union { v16bf v; v8bf h[2]; } au;
    au.v = (v16bf){};
    if (colSel == 0) {                       // only matrix-row 0 of A is the vector
      au.h[0] = *(const v8bf*)(xv + kk + aoff);
      au.h[1] = *(const v8bf*)(xv + kk + aoff + 8);
    }
    v16bf b = *(const v16bf*)(Wrow + kk);
    acc = __builtin_amdgcn_wmma_f32_16x16x32_bf16(false, au.v, false, b,
                                                  (short)0, acc, false, false);
  }
  if (lane < 16) partial[kchunk * G4 + r0 + lane] = acc[0];
}

// ------------- deterministic RNG (splitmix64) for Gumbel-max sampling -------------
__device__ __forceinline__ unsigned long long smix(unsigned long long& s) {
  s += 0x9E3779B97F4A7C15ull;
  unsigned long long z = s;
  z = (z ^ (z >> 30)) * 0xBF58476D1CE4E5B9ull;
  z = (z ^ (z >> 27)) * 0x94D049BB133111EBull;
  return z ^ (z >> 31);
}

// ------------- fused: gate reduce + LSTM cell + logits + sample + next input -------------
__global__ __launch_bounds__(256) void lstm_step(const float* __restrict__ partial,
                                                 const float* __restrict__ b_ih,
                                                 const float* __restrict__ b_hh,
                                                 float* __restrict__ cstate,
                                                 unsigned short* __restrict__ xh,
                                                 const float* __restrict__ w_soft,
                                                 const float* __restrict__ encoder,
                                                 float* __restrict__ out,
                                                 int step) {
  __shared__ float sred[256];
  __shared__ float s_logits[NC];
  __shared__ int   s_op;
  const int t = threadIdx.x;

  auto gate = [&](int r) -> float {
    return partial[r] + partial[G4 + r] + partial[2 * G4 + r] + partial[3 * G4 + r]
         + b_ih[r] + b_hh[r];
  };

  float hreg[HSZ / 256];
  for (int q = 0; q < HSZ / 256; ++q) {
    int j = t + 256 * q;
    float gi = gate(j);
    float gf = gate(HSZ + j);
    float gg = gate(2 * HSZ + j);
    float go = gate(3 * HSZ + j);
    float c2 = sigmoidf(gf) * cstate[j] + sigmoidf(gi) * tanhf(gg);
    float h2 = sigmoidf(go) * tanhf(c2);
    cstate[j]     = c2;
    hreg[q]       = h2;
    xh[HSZ + j]   = f2bf(h2);            // h feeds next step's GEMV (bf16)
  }
  __syncthreads();

  // logits[r] = h2 . w_soft[r]  (12 block reductions over 2048)
  for (int r = 0; r < NC; ++r) {
    float p = 0.0f;
    for (int q = 0; q < HSZ / 256; ++q)
      p += hreg[q] * w_soft[r * HSZ + t + 256 * q];
    sred[t] = p;
    __syncthreads();
    for (int s = 128; s > 0; s >>= 1) {
      if (t < s) sred[t] += sred[t + s];
      __syncthreads();
    }
    if (t == 0) s_logits[r] = sred[0];
    __syncthreads();
  }

  if (t == 0) {
    float l[NC];
    float m = -3.0e38f;
    for (int r = 0; r < NC; ++r) {
      l[r] = 2.5f * tanhf(s_logits[r] * (1.0f / 5.0f));  // TANH_CONSTANT * tanh(logits/T)
      m = fmaxf(m, l[r]);
    }
    float se = 0.0f;
    for (int r = 0; r < NC; ++r) se += __expf(l[r] - m);
    float lse = m + __logf(se);

    // Gumbel-max categorical sample (deterministic surrogate for threefry)
    unsigned long long s = 0x42ull * 0x9E3779B97F4A7C15ull
                         + (unsigned long long)(step + 1) * 0xD1B54A32D192ED03ull;
    int best = 0; float bestv = -3.0e38f;
    for (int r = 0; r < NC; ++r) {
      float u = ((float)(smix(s) >> 40) + 0.5f) * (1.0f / 16777216.0f);
      float v = l[r] - __logf(-__logf(u));
      if (v > bestv) { bestv = v; best = r; }
    }
    float nll = -(l[best] - lse);
    float ent = 0.0f;
    for (int r = 0; r < NC; ++r) {
      float lp = l[r] - lse;
      ent -= __expf(lp) * lp;
    }
    out[step]        = (float)best;
    out[NLAYERS]    += nll;
    out[NLAYERS + 1] += ent;
    s_op = best;
  }
  __syncthreads();

  // next input: x = encoder[op+1] (clamped gather, as in jnp)
  int nxt = min(s_op + 1, NC - 1);
  for (int q = 0; q < HSZ / 256; ++q) {
    int j = t + 256 * q;
    xh[j] = f2bf(encoder[nxt * HSZ + j]);
  }
}

extern "C" void kernel_launch(void* const* d_in, const int* in_sizes, int n_in,
                              void* d_out, int out_size, void* d_ws, size_t ws_size,
                              hipStream_t stream) {
  const float* prev_c  = (const float*)d_in[0];
  const float* prev_h  = (const float*)d_in[1];
  const float* encoder = (const float*)d_in[2];
  const float* w_ih    = (const float*)d_in[3];
  const float* w_hh    = (const float*)d_in[4];
  const float* b_ih    = (const float*)d_in[5];
  const float* b_hh    = (const float*)d_in[6];
  const float* w_soft  = (const float*)d_in[7];
  float* out = (float*)d_out;

  // workspace layout (~67.3 MB): bf16 weights stay resident in 192MB L2 across steps
  const size_t WN = (size_t)G4 * HSZ;                     // 16,777,216 elems per matrix
  unsigned short* wihB = (unsigned short*)d_ws;
  unsigned short* whhB = wihB + WN;
  unsigned short* xh   = whhB + WN;                       // 4096 bf16 = concat(x, h)
  float* partial = (float*)(xh + KTOT);                   // [KSPLIT][8192] f32
  float* cstate  = partial + (size_t)KSPLIT * G4;         // 2048 f32

  cvt_weights<<<(unsigned)(WN / 256), 256, 0, stream>>>(w_ih, w_hh, wihB, whhB);
  init_state<<<1, 256, 0, stream>>>(prev_c, prev_h, encoder, cstate, xh, out);

  for (int step = 0; step < NLAYERS; ++step) {
    // 2048 waves: 512 row-tiles x 4 K-chunks
    gemv_wmma<<<256, 256, 0, stream>>>(wihB, whhB, xh, partial);
    lstm_step<<<1, 256, 0, stream>>>(partial, b_ih, b_hh, cstate, xh,
                                     w_soft, encoder, out, step);
  }
}
